// LightningAttention_3418793968185
// MI455X (gfx1250) — matrix operand order repaired and checked by reading, not compiled, not run on hardware
//
#include <hip/hip_runtime.h>

// ---------------------------------------------------------------------------
// Types
// ---------------------------------------------------------------------------
typedef __attribute__((ext_vector_type(16))) __bf16 bf16x16;
typedef __attribute__((ext_vector_type(8)))  __bf16 bf16x8;
typedef __attribute__((ext_vector_type(4)))  __bf16 bf16x4;
typedef __attribute__((ext_vector_type(8)))  float  v8f;
// exact parameter type of __builtin_amdgcn_global_load_async_to_lds_b128
typedef __attribute__((__vector_size__(4 * sizeof(int)))) int gvec4i;

#define B_  2
#define T_  4096
#define HID_ 2048
#define H_  16
#define D_  128
#define C_  64
#define NC_ (T_ / C_)
#define DSPL_ 4               // dv-split of the GLA state scan
#define DVW_ (D_ / DSPL_)     // 32 dv columns per workgroup
#define SCALE_ 0.08838834764831845f  // 1/sqrt(128)

// ---------------------------------------------------------------------------
// WMMA fragment helpers (CDNA5 wave32 layouts, CDNA5 ISA §7.12.2)
//  A (16x32, 16-bit): lane L -> m = L&15 ; elems 0..7 = K=8*(L>>4)+j,
//                     elems 8..15 = K=16+8*(L>>4)+j  (two 16B runs)
//  B (32x16, 16-bit): lane L -> n = L&15 ; same K order as A (two 16B runs)
//  C/D (16x16 f32):   lane L -> n = L&15 ; reg r -> m = r + 8*(L>>4)
// ---------------------------------------------------------------------------
__device__ __forceinline__ bf16x16 load_frag_a(const __bf16* base, int stride,
                                               int m0, int k0, int lane) {
  const __bf16* p = base + (m0 + (lane & 15)) * stride + k0 + ((lane >> 4) << 3);
  bf16x8 lo = *(const bf16x8*)p;
  bf16x8 hi = *(const bf16x8*)(p + 16);
  bf16x16 r;
#pragma unroll
  for (int i = 0; i < 8; ++i) { r[i] = lo[i]; r[i + 8] = hi[i]; }
  return r;
}

// base is stored [N][K] (i.e. value for B(k,n) lives at base[n*stride + k])
__device__ __forceinline__ bf16x16 load_frag_b(const __bf16* base, int stride,
                                               int k0, int n0, int lane) {
  const __bf16* p = base + (n0 + (lane & 15)) * stride + k0 + ((lane >> 4) << 3);
  bf16x8 lo = *(const bf16x8*)p;
  bf16x8 hi = *(const bf16x8*)(p + 16);
  bf16x16 r;
#pragma unroll
  for (int i = 0; i < 8; ++i) { r[i] = lo[i]; r[i + 8] = hi[i]; }
  return r;
}

__device__ __forceinline__ v8f wmma_bf16(bf16x16 a, bf16x16 b, v8f c) {
  return __builtin_amdgcn_wmma_f32_16x16x32_bf16(false, a, false, b,
                                                 (short)0, c, false, false);
}

__device__ __forceinline__ v8f v8f_zero() {
  v8f z = {0.f, 0.f, 0.f, 0.f, 0.f, 0.f, 0.f, 0.f};
  return z;
}

// ---------------------------------------------------------------------------
// Async global->LDS copy (GLOBAL_LOAD_ASYNC_TO_LDS_B128, ASYNCcnt), with a
// synchronous fallback if the builtin is unavailable on this toolchain.
// ---------------------------------------------------------------------------
#if __has_builtin(__builtin_amdgcn_global_load_async_to_lds_b128) && \
    __has_builtin(__builtin_amdgcn_s_wait_asynccnt)
#define HAVE_ASYNC_LDS 1
#else
#define HAVE_ASYNC_LDS 0
#endif

__device__ __forceinline__ void async_copy_b128(const __bf16* g, __bf16* l) {
#if HAVE_ASYNC_LDS
  __builtin_amdgcn_global_load_async_to_lds_b128(
      (__attribute__((address_space(1))) gvec4i*)(unsigned long)g,
      (__attribute__((address_space(3))) gvec4i*)(unsigned long)l, 0, 0);
#else
  *(bf16x8*)l = *(const bf16x8*)g;
#endif
}

template <int N>
__device__ __forceinline__ void async_wait() {
#if HAVE_ASYNC_LDS
  __builtin_amdgcn_s_wait_asynccnt(N);
#endif
}

// ---------------------------------------------------------------------------
// fp32 -> bf16 elementwise convert (vector of 4)
// ---------------------------------------------------------------------------
__global__ __launch_bounds__(256) void f32_to_bf16_kernel(
    const float* __restrict__ in, __bf16* __restrict__ out, long n4) {
  long i = (long)blockIdx.x * blockDim.x + threadIdx.x;
  if (i >= n4) return;
  long i4 = i * 4;
  float4 v = *(const float4*)(in + i4);
  bf16x4 o;
  o[0] = (__bf16)v.x; o[1] = (__bf16)v.y; o[2] = (__bf16)v.z; o[3] = (__bf16)v.w;
  *(bf16x4*)(out + i4) = o;
}

// ---------------------------------------------------------------------------
// transpose-convert: in fp32 [R][Ccol] -> out bf16 [Ccol][R]
// ---------------------------------------------------------------------------
__global__ __launch_bounds__(256) void transpose_f32_to_bf16(
    const float* __restrict__ in, __bf16* __restrict__ out, int R, int Ccol) {
  __shared__ float tile[32][33];
  int c0 = blockIdx.x * 32, r0 = blockIdx.y * 32;
  int tx = threadIdx.x, ty = threadIdx.y;  // block (32,8)
#pragma unroll
  for (int i = 0; i < 32; i += 8)
    tile[ty + i][tx] = in[(long)(r0 + ty + i) * Ccol + (c0 + tx)];
  __syncthreads();
#pragma unroll
  for (int i = 0; i < 32; i += 8)
    out[(long)(c0 + ty + i) * R + (r0 + tx)] = (__bf16)tile[tx][ty + i];
}

// ---------------------------------------------------------------------------
// Tiled bf16 WMMA GEMM: C[M,N] = A[M,K] * B  (B given transposed: Bt[N][K])
// 256x128 block tile, 256 threads (8 waves), wave owns 4x4 16x16 tiles.
// K panels of 64 double-buffered via GLOBAL_LOAD_ASYNC_TO_LDS_B128:
// panel p+1 streams in (12 outstanding per wave) while panel p is consumed
// by 32 WMMAs/wave; s_wait_asynccnt 12 releases as soon as panel p landed.
// ---------------------------------------------------------------------------
#define NPER 12  // async b128 copies issued per thread per panel (8 A + 4 B)

template <bool OUTF32>
__global__ __launch_bounds__(256) void gemm_bf16_wmma(
    const __bf16* __restrict__ A, const __bf16* __restrict__ Bt,
    void* __restrict__ Cv, int M, int N, int K) {
  __shared__ __bf16 sA[2][256 * 72];
  __shared__ __bf16 sB[2][128 * 72];
  const int tid = threadIdx.x, wave = tid >> 5, lane = tid & 31;
  const int m0 = blockIdx.x * 256, n0 = blockIdx.y * 128;
  const int wm = (wave & 3) * 4;   // m-tile base (of 16)
  const int wn = (wave >> 2) * 4;  // n-tile base (of 8)

  v8f acc[4][4];
#pragma unroll
  for (int i = 0; i < 4; ++i)
#pragma unroll
    for (int j = 0; j < 4; ++j) acc[i][j] = v8f_zero();

  // staging assignment: A row per thread (64 cols), B row = tid&127 (32 cols)
  const __bf16* Ap = A + (long)(m0 + tid) * K;
  const int rB = tid & 127, cB = (tid >> 7) * 32;
  const __bf16* Bp = Bt + (long)(n0 + rB) * K + cB;
  const int aOff = tid * 72, bOff = rB * 72 + cB;

  auto issue_panel = [&](int ks, int buf) {
#pragma unroll
    for (int c = 0; c < 8; ++c)
      async_copy_b128(Ap + ks + c * 8, &sA[buf][aOff + c * 8]);
#pragma unroll
    for (int c = 0; c < 4; ++c)
      async_copy_b128(Bp + ks + c * 8, &sB[buf][bOff + c * 8]);
  };

  const int P = K / 64;
  issue_panel(0, 0);  // prologue: panel 0 -> buffer 0

  for (int p = 0; p < P; ++p) {
    const int cur = p & 1;
    if (p + 1 < P) {
      issue_panel((p + 1) * 64, 1 - cur);  // stream next panel
      __builtin_prefetch(Ap + (p + 1) * 64, 0, 1);  // global_prefetch_b8
      async_wait<NPER>();  // panel p landed; panel p+1 still in flight
    } else {
      async_wait<0>();
    }
    __syncthreads();  // all waves see panel p in LDS

#pragma unroll
    for (int kk = 0; kk < 64; kk += 32) {
      bf16x16 af[4];
#pragma unroll
      for (int i = 0; i < 4; ++i)
        af[i] = load_frag_a(sA[cur], 72, (wm + i) * 16, kk, lane);
#pragma unroll
      for (int j = 0; j < 4; ++j) {
        bf16x16 bf_ = load_frag_b(sB[cur], 72, kk, (wn + j) * 16, lane);
#pragma unroll
        for (int i = 0; i < 4; ++i) acc[i][j] = wmma_bf16(af[i], bf_, acc[i][j]);
      }
    }
    __syncthreads();  // buf[cur] free before iter p+1 streams into it
  }

  const int nloc = lane & 15, mh = (lane >> 4) * 8;
#pragma unroll
  for (int i = 0; i < 4; ++i)
#pragma unroll
    for (int j = 0; j < 4; ++j)
#pragma unroll
      for (int rr = 0; rr < 8; ++rr) {
        int gm = m0 + (wm + i) * 16 + mh + rr;
        int gn = n0 + (wn + j) * 16 + nloc;
        if (OUTF32)
          ((float*)Cv)[(long)gm * N + gn] = acc[i][j][rr];
        else
          ((__bf16*)Cv)[(long)gm * N + gn] = (__bf16)acc[i][j][rr];
      }
}

// ---------------------------------------------------------------------------
// In-place RMSNorm over D=128 (wave per row; 32 lanes x 4 values)
// grid.y == 0 -> q/qn_w, grid.y == 1 -> k/kn_w
// ---------------------------------------------------------------------------
__global__ __launch_bounds__(256) void rmsnorm_kernel(
    __bf16* __restrict__ q, __bf16* __restrict__ k,
    const float* __restrict__ qw, const float* __restrict__ kw, long rows) {
  const int wave = threadIdx.x >> 5, lane = threadIdx.x & 31;
  long row = (long)blockIdx.x * 8 + wave;
  if (row >= rows) return;
  __bf16* buf = (blockIdx.y == 0) ? q : k;
  const float* w = (blockIdx.y == 0) ? qw : kw;

  bf16x4 v = *(const bf16x4*)(buf + row * 128 + lane * 4);
  float f0 = (float)v[0], f1 = (float)v[1], f2 = (float)v[2], f3 = (float)v[3];
  float ss = f0 * f0 + f1 * f1 + f2 * f2 + f3 * f3;
#pragma unroll
  for (int off = 16; off >= 1; off >>= 1) ss += __shfl_xor(ss, off, 32);
  float rr = rsqrtf(ss * (1.0f / 128.0f) + 1e-6f);
  bf16x4 o;
  o[0] = (__bf16)(f0 * rr * w[lane * 4 + 0]);
  o[1] = (__bf16)(f1 * rr * w[lane * 4 + 1]);
  o[2] = (__bf16)(f2 * rr * w[lane * 4 + 2]);
  o[3] = (__bf16)(f3 * rr * w[lane * 4 + 3]);
  *(bf16x4*)(buf + row * 128 + lane * 4) = o;
}

// ---------------------------------------------------------------------------
// Chunked simple-GLA scan, dv-split by DSPL_. One workgroup per (b,h,dvs);
// 8 waves. The workgroup owns state slice S^T[dv0..dv0+31][0..127] in fp32
// WMMA accumulators (2 tiles per wave). Intra-chunk attn is recomputed per
// slice (cheap: +25% attn flops for 4x the parallelism).
// ---------------------------------------------------------------------------
#define LW 136  // D + 8 pad (rows 16B aligned)
#define LT 72   // C + 8 pad

__global__ __launch_bounds__(256) void gla_chunk_kernel(
    const __bf16* __restrict__ qg, const __bf16* __restrict__ kg,
    const __bf16* __restrict__ vg, __bf16* __restrict__ og) {
  const int idx = blockIdx.x;      // B_*H_*DSPL_ = 128
  const int dvs = idx & (DSPL_ - 1);
  const int h = (idx >> 2) & (H_ - 1);
  const int b = idx >> 6;
  const int tid = threadIdx.x, wave = tid >> 5, lane = tid & 31;

  // ALiBi slope for H=16 (power of 2): slope_h = 2^(-0.5*(h+1)); g = -slope
  const float g = -exp2f(-0.5f * (float)(h + 1));
  const float gC = expf(g * (float)C_);

  __shared__ __bf16 sQ[C_ * LW];      // q[j][d]
  __shared__ __bf16 sQ2[C_ * LW];     // q[j][d] * SCALE*exp(g*(j+1))
  __shared__ __bf16 sK[C_ * LW];      // k[j][d]
  __shared__ __bf16 sKdT[D_ * LT];    // (k*decay_k)[j][dk] stored [dk][j]
  __shared__ __bf16 sVT[DVW_ * LT];   // v[j][dv] stored [dv_local][j]
  __shared__ __bf16 sA[C_ * LT];      // masked attn bf16 [i][j]
  __shared__ __bf16 sST[DVW_ * LW];   // S^T bf16: sST[dv_local][dk]

  // S^T accumulator tiles: dv-tile tv = wave>>2 (0..1), dk-tiles tk0,tk0+1
  const int tv = wave >> 2;
  const int tk0 = (wave & 3) * 2;
  v8f st[2];
  st[0] = v8f_zero();
  st[1] = v8f_zero();

  const long baseBH = (long)b * T_ * (H_ * D_) + (long)h * D_;
  const int nloc = lane & 15, mh = (lane >> 4) * 8;

  for (int nc = 0; nc < NC_; ++nc) {
    __syncthreads();  // previous chunk fully consumed before LDS overwrite

    // ---- stage q,k (full 64x128) ----
#pragma unroll
    for (int it = 0; it < 4; ++it) {
      int vdx = tid + it * 256;        // 0..1023
      int j = vdx >> 4;                // row in chunk
      int d8 = (vdx & 15) * 8;         // col base
      long gidx = baseBH + (long)(nc * C_ + j) * (H_ * D_) + d8;
      bf16x8 qv = *(const bf16x8*)(qg + gidx);
      bf16x8 kv = *(const bf16x8*)(kg + gidx);
      *(bf16x8*)(sQ + j * LW + d8) = qv;
      *(bf16x8*)(sK + j * LW + d8) = kv;
      float dq = SCALE_ * expf(g * (float)(j + 1));
      float dk = expf(g * (float)(C_ - 1 - j));
      bf16x8 q2;
#pragma unroll
      for (int e = 0; e < 8; ++e) {
        q2[e] = (__bf16)((float)qv[e] * dq);
        sKdT[(d8 + e) * LT + j] = (__bf16)((float)kv[e] * dk);
      }
      *(bf16x8*)(sQ2 + j * LW + d8) = q2;
    }
    // ---- stage v slice (64 x 32), transposed into sVT[dv_local][j] ----
    {
      int j = tid >> 2;                // 0..63
      int d8l = (tid & 3) * 8;         // 0..24
      long gidx = baseBH + (long)(nc * C_ + j) * (H_ * D_) + dvs * DVW_ + d8l;
      bf16x8 vv = *(const bf16x8*)(vg + gidx);
#pragma unroll
      for (int e = 0; e < 8; ++e) sVT[(d8l + e) * LT + j] = vv[e];
    }
    __syncthreads();

    // ---- 1) attn = q @ k^T  (16 tiles, 2 per wave), mask+scale -> sA ----
    {
      const int mi = wave >> 1;
      const int nj0 = (wave & 1) * 2;
      v8f att0 = v8f_zero(), att1 = v8f_zero();
#pragma unroll
      for (int ks = 0; ks < 4; ++ks) {
        bf16x16 a = load_frag_a(sQ, LW, mi * 16, ks * 32, lane);
        bf16x16 b0 = load_frag_b(sK, LW, ks * 32, nj0 * 16, lane);
        bf16x16 b1 = load_frag_b(sK, LW, ks * 32, (nj0 + 1) * 16, lane);
        att0 = wmma_bf16(a, b0, att0);
        att1 = wmma_bf16(a, b1, att1);
      }
#pragma unroll
      for (int j = 0; j < 2; ++j) {
        const v8f& at = j ? att1 : att0;
        int jj = (nj0 + j) * 16 + nloc;
#pragma unroll
        for (int r2 = 0; r2 < 8; ++r2) {
          int i = mi * 16 + mh + r2;
          float f = (i >= jj) ? SCALE_ * expf(g * (float)(i - jj)) : 0.0f;
          sA[i * LT + jj] = (__bf16)(at[r2] * f);
        }
      }
    }

    // ---- 2) dump S^T slice (pre-update) to LDS as bf16 ----
#pragma unroll
    for (int ti = 0; ti < 2; ++ti)
#pragma unroll
      for (int r2 = 0; r2 < 8; ++r2)
        sST[(tv * 16 + mh + r2) * LW + (tk0 + ti) * 16 + nloc] =
            (__bf16)st[ti][r2];
    __syncthreads();

    // ---- 3) o = attn @ v + q2 @ S  (8 tiles, 1 per wave) ----
    {
      const int mo = wave & 3;   // m-tile (i)
      const int nv = wave >> 2;  // dv-tile (local)
      v8f oc = v8f_zero();
#pragma unroll
      for (int ks = 0; ks < 2; ++ks) {  // intra: K = C = 64
        bf16x16 a = load_frag_a(sA, LT, mo * 16, ks * 32, lane);
        bf16x16 bb = load_frag_b(sVT, LT, ks * 32, nv * 16, lane);
        oc = wmma_bf16(a, bb, oc);
      }
#pragma unroll
      for (int ks = 0; ks < 4; ++ks) {  // inter: K = D = 128
        bf16x16 a = load_frag_a(sQ2, LW, mo * 16, ks * 32, lane);
        bf16x16 bb = load_frag_b(sST, LW, ks * 32, nv * 16, lane);
        oc = wmma_bf16(a, bb, oc);
      }
      int dvg = dvs * DVW_ + nv * 16 + nloc;
#pragma unroll
      for (int r2 = 0; r2 < 8; ++r2) {
        int i = mo * 16 + mh + r2;
        og[baseBH + (long)(nc * C_ + i) * (H_ * D_) + dvg] = (__bf16)oc[r2];
      }
    }

    // ---- 4) S^T = gC * S^T + v^T @ (k*decay_k) ----
    {
      st[0] = st[0] * gC;
      st[1] = st[1] * gC;
#pragma unroll
      for (int ks = 0; ks < 2; ++ks) {  // K = C = 64
        bf16x16 a = load_frag_a(sVT, LT, tv * 16, ks * 32, lane);
#pragma unroll
        for (int ti = 0; ti < 2; ++ti) {
          bf16x16 bb = load_frag_b(sKdT, LT, ks * 32, (tk0 + ti) * 16, lane);
          st[ti] = wmma_bf16(a, bb, st[ti]);
        }
      }
    }
  }
}

// ---------------------------------------------------------------------------
// Host-side pipeline
// ---------------------------------------------------------------------------
extern "C" void kernel_launch(void* const* d_in, const int* in_sizes, int n_in,
                              void* d_out, int out_size, void* d_ws,
                              size_t ws_size, hipStream_t stream) {
  const float* x   = (const float*)d_in[0];  // [B,T,HID]
  const float* Wq  = (const float*)d_in[1];  // [HID, H*D]
  const float* Wk  = (const float*)d_in[2];
  const float* Wv  = (const float*)d_in[3];
  const float* Wo  = (const float*)d_in[4];  // [H*D, HID]
  const float* qnw = (const float*)d_in[5];  // [D]
  const float* knw = (const float*)d_in[6];  // [D]

  const long MROWS = (long)B_ * T_;       // 8192
  const long ACT   = MROWS * (H_ * D_);   // 16,777,216 elements
  const long WELEM = (long)HID_ * (H_ * D_);

  __bf16* xb  = (__bf16*)d_ws;
  __bf16* wqT = xb + ACT;
  __bf16* wkT = wqT + WELEM;
  __bf16* wvT = wkT + WELEM;
  __bf16* woT = wvT + WELEM;
  __bf16* qb  = woT + WELEM;
  __bf16* kb  = qb + ACT;
  __bf16* vb  = kb + ACT;
  __bf16* ob  = vb + ACT;

  // 1) x -> bf16
  {
    long n4 = (MROWS * HID_) / 4;
    f32_to_bf16_kernel<<<dim3((unsigned)((n4 + 255) / 256)), 256, 0, stream>>>(
        x, xb, n4);
  }
  // 2) weights -> transposed bf16 [N][K]
  {
    dim3 tg(HID_ / 32, HID_ / 32), tb(32, 8);
    transpose_f32_to_bf16<<<tg, tb, 0, stream>>>(Wq, wqT, HID_, H_ * D_);
    transpose_f32_to_bf16<<<tg, tb, 0, stream>>>(Wk, wkT, HID_, H_ * D_);
    transpose_f32_to_bf16<<<tg, tb, 0, stream>>>(Wv, wvT, HID_, H_ * D_);
    transpose_f32_to_bf16<<<tg, tb, 0, stream>>>(Wo, woT, H_ * D_, HID_);
  }
  // 3) projections q,k,v = x @ W (bf16 out)
  {
    dim3 gg((unsigned)(MROWS / 256), (H_ * D_) / 128);
    gemm_bf16_wmma<false><<<gg, 256, 0, stream>>>(xb, wqT, qb, (int)MROWS,
                                                  H_ * D_, HID_);
    gemm_bf16_wmma<false><<<gg, 256, 0, stream>>>(xb, wkT, kb, (int)MROWS,
                                                  H_ * D_, HID_);
    gemm_bf16_wmma<false><<<gg, 256, 0, stream>>>(xb, wvT, vb, (int)MROWS,
                                                  H_ * D_, HID_);
  }
  // 4) RMSNorm(q), RMSNorm(k) in place
  {
    long rows = MROWS * H_;  // 131072
    rmsnorm_kernel<<<dim3((unsigned)(rows / 8), 2), 256, 0, stream>>>(
        qb, kb, qnw, knw, rows);
  }
  // 5) chunked GLA scan (dv-split x4 -> 128 workgroups)
  gla_chunk_kernel<<<dim3(B_ * H_ * DSPL_), 256, 0, stream>>>(qb, kb, vb, ob);

  // 6) out = o @ Wo (fp32 out)
  {
    dim3 gg((unsigned)(MROWS / 256), HID_ / 128);
    gemm_bf16_wmma<true><<<gg, 256, 0, stream>>>(ob, woT, d_out, (int)MROWS,
                                                 HID_, H_ * D_);
  }
}
